// VariLengthInputLayer_25245817766348
// MI455X (gfx1250) — matrix operand order repaired and checked by reading, not compiled, not run on hardware
//
#include <hip/hip_runtime.h>
#include <hip/hip_bf16.h>

// ---------------------------------------------------------------------------
// Types for CDNA5 WMMA (wave32, v_wmma_f32_16x16x32_f16)
// ---------------------------------------------------------------------------
typedef __attribute__((ext_vector_type(16))) _Float16 v16h;
typedef __attribute__((ext_vector_type(8)))  float    v8f;

union Frag {
    v16h h;
    uint4 u[2];
};

struct __align__(8) h4 { _Float16 x, y, z, w; };

__device__ inline h4 ld4h(const float* p) {
    float4 f = *(const float4*)p;
    h4 r;
    r.x = (_Float16)f.x; r.y = (_Float16)f.y;
    r.z = (_Float16)f.z; r.w = (_Float16)f.w;
    return r;
}
__device__ inline h4 ld4h(const _Float16* p) {
    return *(const h4*)p;
}

// ---------------------------------------------------------------------------
// Tiled WMMA GEMM:  C[M x N] = A[M x K] * B[N x K]^T   (both K-contiguous)
//   A: TA (float or _Float16), converted to f16 while staging to LDS
//   B: float weights W[o, d] row-major (K contiguous), converted to f16
//   C: fp32
// Tile: BM=128, BN=128, BK=64; 256 threads = 8 waves (2x4 wave grid),
// each wave owns a 64x32 C tile = 4x2 accumulators of 16x16.
// Requires M%128==0, N%128==0, K%64==0 (true for this problem).
// ---------------------------------------------------------------------------
#define LDSW 72  // halves per LDS tile row (64 data + 8 pad), 144B: 16B aligned

template <typename TA>
__global__ __launch_bounds__(256) void gemm_wmma_kernel(
    const TA* __restrict__ A, int lda,
    const float* __restrict__ B, int ldb,
    float* __restrict__ C, int ldc, int K)
{
    __shared__ __align__(16) _Float16 sA[128 * LDSW];
    __shared__ __align__(16) _Float16 sB[128 * LDSW];

    const int t    = threadIdx.x;
    const int lid  = t & 31;
    const int wid  = t >> 5;
    const int wm   = wid >> 2;      // 0..1
    const int wn   = wid & 3;       // 0..3
    const int lrow = lid & 15;      // row within 16x16 tile
    const int hsel = lid >> 4;      // half-select (lane group)

    const size_t m0 = (size_t)blockIdx.y * 128;
    const size_t n0 = (size_t)blockIdx.x * 128;

    // staging coords: 16 threads per row (4 elems each), 16 rows per pass, 8 passes
    const int sr = t >> 4;          // 0..15
    const int sc = (t & 15) * 4;    // 0..60

    v8f acc[4][2];
#pragma unroll
    for (int sm = 0; sm < 4; ++sm)
#pragma unroll
        for (int sn = 0; sn < 2; ++sn)
            acc[sm][sn] = (v8f){0.f,0.f,0.f,0.f,0.f,0.f,0.f,0.f};

    for (int kt = 0; kt < K; kt += 64) {
        const TA*    Ag = A + (m0 + sr) * (size_t)lda + kt + sc;
        const float* Bg = B + (n0 + sr) * (size_t)ldb + kt + sc;
#pragma unroll
        for (int i = 0; i < 8; ++i) {
            *(h4*)(&sA[(sr + i * 16) * LDSW + sc]) = ld4h(Ag + (size_t)(i * 16) * lda);
            *(h4*)(&sB[(sr + i * 16) * LDSW + sc]) = ld4h(Bg + (size_t)(i * 16) * ldb);
        }
        if (kt + 64 < K) {  // CDNA5 global_prefetch_b8 for the next K-tile
            __builtin_prefetch(Ag + 64, 0, 1);
            __builtin_prefetch(Bg + 64, 0, 1);
        }
        __syncthreads();

#pragma unroll
        for (int kk = 0; kk < 64; kk += 32) {
            Frag fa[4], fb[2];
#pragma unroll
            for (int sm = 0; sm < 4; ++sm) {
                // ISA 7.12.2 A layout: lanes 0-15 hold K {0..7,16..23},
                // lanes 16-31 hold K {8..15,24..31} of row M=lane%16
                const _Float16* p = &sA[(wm * 64 + sm * 16 + lrow) * LDSW + kk + hsel * 8];
                fa[sm].u[0] = *(const uint4*)(p);
                fa[sm].u[1] = *(const uint4*)(p + 16);
            }
#pragma unroll
            for (int sn = 0; sn < 2; ++sn) {
                // B layout: lane n holds column N=n%16, K {0..7,16..23} (lanes 0-15) /
                // K {8..15,24..31} (lanes 16-31), packed 2 per VGPR
                const _Float16* p = &sB[(wn * 32 + sn * 16 + lrow) * LDSW + kk + hsel * 8];
                fb[sn].u[0] = *(const uint4*)(p);
                fb[sn].u[1] = *(const uint4*)(p + 16);
            }
#pragma unroll
            for (int sm = 0; sm < 4; ++sm)
#pragma unroll
                for (int sn = 0; sn < 2; ++sn)
                    acc[sm][sn] = __builtin_amdgcn_wmma_f32_16x16x32_f16(
                        false, fa[sm].h, false, fb[sn].h,
                        (short)0, acc[sm][sn], false, false);
        }
        __syncthreads();
    }

    // 32-bit C/D layout: lane l -> col n = l%16; VGPR r -> row r + 8*(l/16)
#pragma unroll
    for (int sm = 0; sm < 4; ++sm) {
#pragma unroll
        for (int sn = 0; sn < 2; ++sn) {
            const size_t col  = n0 + wn * 32 + sn * 16 + lrow;
            const size_t row0 = m0 + wm * 64 + sm * 16 + hsel * 8;
#pragma unroll
            for (int r = 0; r < 8; ++r)
                C[(row0 + r) * (size_t)ldc + col] = acc[sm][sn][r];
        }
    }
}

// ---------------------------------------------------------------------------
// Attention (M=4 tokens, NH=16 heads, DK=DV=64) + ctx->f16 for the FC GEMM.
// One 64-thread block per batch row; q/k/v[4x1024] staged in LDS.
// Thread (h, qm): h = tid/4, qm = tid%4.
// ---------------------------------------------------------------------------
__global__ __launch_bounds__(64) void attn_kernel(
    const float* __restrict__ q, const float* __restrict__ k,
    const float* __restrict__ v, float* __restrict__ attn_out,
    _Float16* __restrict__ ctx)
{
    __shared__ float sq[4096];
    __shared__ float sk[4096];
    __shared__ float sv[4096];

    const int n   = blockIdx.x;
    const int tid = threadIdx.x;

    const float* qg = q + (size_t)n * 4096;
    const float* kg = k + (size_t)n * 4096;
    const float* vg = v + (size_t)n * 4096;
    for (int i = tid; i < 4096; i += 64) {
        sq[i] = qg[i];
        sk[i] = kg[i];
        sv[i] = vg[i];
    }
    __syncthreads();

    const int h  = tid >> 2;   // 0..15
    const int qm = tid & 3;    // 0..3
    const int hb = h * 64;

    float qv[64];
#pragma unroll
    for (int d = 0; d < 64; ++d) qv[d] = sq[qm * 1024 + hb + d];

    float s[4];
#pragma unroll
    for (int km = 0; km < 4; ++km) {
        float acc = 0.f;
#pragma unroll
        for (int d = 0; d < 64; ++d) acc += qv[d] * sk[km * 1024 + hb + d];
        s[km] = acc * 0.125f;  // 1/sqrt(64)
    }

    float mx = fmaxf(fmaxf(s[0], s[1]), fmaxf(s[2], s[3]));
    float p[4], sum = 0.f;
#pragma unroll
    for (int km = 0; km < 4; ++km) { p[km] = __expf(s[km] - mx); sum += p[km]; }
    const float inv = 1.f / sum;
#pragma unroll
    for (int km = 0; km < 4; ++km) p[km] *= inv;

    // attn: [bs, NH, 4, 4]
    float* ao = attn_out + ((((size_t)n * 16 + h) * 4 + qm) * 4);
#pragma unroll
    for (int km = 0; km < 4; ++km) ao[km] = p[km];

    // ctx: [bs, M, NH*DV] row (n*4+qm), cols hb..hb+63
    _Float16* cp = ctx + ((size_t)n * 4 + qm) * 1024 + hb;
#pragma unroll
    for (int d = 0; d < 64; ++d) {
        float c = 0.f;
#pragma unroll
        for (int km = 0; km < 4; ++km) c += p[km] * sv[km * 1024 + hb + d];
        cp[d] = (_Float16)c;
    }
}

// ---------------------------------------------------------------------------
// Fused epilogue: out = LN(fc_out + fc_b + residual_v) * g + b, in-place.
// One 256-thread block per row of 1024.
// ---------------------------------------------------------------------------
__global__ __launch_bounds__(256) void ln_kernel(
    float* __restrict__ out, const float* __restrict__ vres,
    const float* __restrict__ fc_b, const float* __restrict__ g,
    const float* __restrict__ b)
{
    __shared__ float rs[256];
    __shared__ float rss[256];

    const size_t row = blockIdx.x;
    const int t = threadIdx.x;
    float* o = out + row * 1024;
    const float* r = vres + row * 1024;

    float vals[4];
    float s = 0.f, ss = 0.f;
#pragma unroll
    for (int i = 0; i < 4; ++i) {
        const int j = t + 256 * i;
        const float x = o[j] + fc_b[j] + r[j];
        vals[i] = x; s += x; ss += x * x;
    }
    rs[t] = s; rss[t] = ss;
    __syncthreads();
    for (int off = 128; off > 0; off >>= 1) {
        if (t < off) { rs[t] += rs[t + off]; rss[t] += rss[t + off]; }
        __syncthreads();
    }
    const float mean = rs[0] * (1.f / 1024.f);
    const float var  = rss[0] * (1.f / 1024.f) - mean * mean;
    const float inv  = rsqrtf(var + 1e-6f);
#pragma unroll
    for (int i = 0; i < 4; ++i) {
        const int j = t + 256 * i;
        o[j] = (vals[i] - mean) * inv * g[j] + b[j];
    }
}

// ---------------------------------------------------------------------------
// Launch
// ---------------------------------------------------------------------------
extern "C" void kernel_launch(void* const* d_in, const int* in_sizes, int n_in,
                              void* d_out, int out_size, void* d_ws, size_t ws_size,
                              hipStream_t stream)
{
    (void)in_sizes; (void)n_in; (void)out_size; (void)ws_size;

    static const int dims[4] = {2048, 1024, 512, 4096};
    static const int offs[4] = {0, 2048, 3072, 3584};
    const int BS = 8192, DM = 1024, LDX = 7680;
    const size_t QKV_ELEMS = (size_t)BS * 4 * DM;  // 33,554,432

    const float* x    = (const float*)d_in[0];
    const float* fc_w = (const float*)d_in[13];
    const float* fc_b = (const float*)d_in[14];
    const float* ln_g = (const float*)d_in[15];
    const float* ln_b = (const float*)d_in[16];

    float* q = (float*)d_ws;
    float* k = q + QKV_ELEMS;
    float* v = k + QKV_ELEMS;
    _Float16* ctx = (_Float16*)(v + QKV_ELEMS);

    float* out      = (float*)d_out;                 // [bs, 4, 1024]
    float* attn_out = out + QKV_ELEMS;               // [bs, 16, 4, 4]

    // QKV projections: 12 WMMA GEMMs, C strided into [bs, M, 1024] layout
    const dim3 gq(DM / 128, BS / 128);
    for (int i = 0; i < 4; ++i) {
        const float* wq = (const float*)d_in[1 + 3 * i];
        const float* wk = (const float*)d_in[2 + 3 * i];
        const float* wv = (const float*)d_in[3 + 3 * i];
        gemm_wmma_kernel<float><<<gq, 256, 0, stream>>>(
            x + offs[i], LDX, wq, dims[i], q + i * DM, 4 * DM, dims[i]);
        gemm_wmma_kernel<float><<<gq, 256, 0, stream>>>(
            x + offs[i], LDX, wk, dims[i], k + i * DM, 4 * DM, dims[i]);
        gemm_wmma_kernel<float><<<gq, 256, 0, stream>>>(
            x + offs[i], LDX, wv, dims[i], v + i * DM, 4 * DM, dims[i]);
    }

    // Attention + softmax + ctx (f16)
    attn_kernel<<<BS, 64, 0, stream>>>(q, k, v, attn_out, ctx);

    // FC: [bs*4, 1024] x [1024, 1024]^T -> d_out (fp32, no bias yet)
    gemm_wmma_kernel<_Float16><<<dim3(DM / 128, (BS * 4) / 128), 256, 0, stream>>>(
        ctx, DM, fc_w, DM, out, DM, DM);

    // bias + residual + LayerNorm, in place
    ln_kernel<<<BS * 4, 256, 0, stream>>>(out, v, fc_b, ln_g, ln_b);
}
